// MoEBlock_17935783428598
// MI455X (gfx1250) — matrix operand order repaired and checked by reading, not compiled, not run on hardware
//
#include <hip/hip_runtime.h>
#include <hip/hip_bf16.h>
#include <math.h>

#define D_MODEL 1024
#define N_EXPERTS 8
#define BN 64
#define T_TOKENS 8192
#define ADAPTER_SCALE_F 0.5f

typedef __attribute__((ext_vector_type(16))) __bf16 v16bf;
typedef __attribute__((ext_vector_type(8)))  __bf16 v8bf;
typedef __attribute__((ext_vector_type(8)))  float  v8f;
typedef __attribute__((ext_vector_type(4)))  float  v4f;
typedef __attribute__((ext_vector_type(2)))  float  v2f;

__device__ __forceinline__ __bf16 to_bf16(float f) { return (__bf16)f; }

// Async global->LDS 16-byte copy (CDNA5, ASYNCcnt-tracked).
// ldsoff = LDS byte offset (low 32 bits of the flat shared address).
__device__ __forceinline__ void async_copy_b128(unsigned ldsoff, const void* gsrc) {
    asm volatile("global_load_async_to_lds_b128 %0, %1, off"
                 :: "v"(ldsoff), "v"(gsrc) : "memory");
}
__device__ __forceinline__ void wait_async0() {
#if __has_builtin(__builtin_amdgcn_s_wait_asynccnt)
    __builtin_amdgcn_s_wait_asynccnt(0);
#else
    asm volatile("s_wait_asynccnt 0x0" ::: "memory");
#endif
}
__device__ __forceinline__ unsigned lds_off(const void* p) {
    return (unsigned)(size_t)p;   // LDS aperture: flat addr low 32 bits = LDS byte addr
}

// ---------------------------------------------------------------------------
// One-time weight preprocessing: f32 -> bf16 with transpose so that staging
// into LDS becomes contiguous 16B copies.
//   Wd_t[e][n][k]  (e<8, n<64,  k<1024) = Wd[e][k][n]
//   Wu_t[n][kk]    (n<1024, kk<512)     = Wu[kk/64][kk%64][n]
// ---------------------------------------------------------------------------
__global__ __launch_bounds__(256) void prep_wd(const float* __restrict__ Wd,
                                               __bf16* __restrict__ Wd_t)
{
    int idx = blockIdx.x * 256 + threadIdx.x;       // 524288 total
    int k = idx & 1023, n = (idx >> 10) & 63, e = idx >> 16;
    Wd_t[idx] = to_bf16(Wd[(size_t)e * D_MODEL * BN + (size_t)k * BN + n]);
}
__global__ __launch_bounds__(256) void prep_wu(const float* __restrict__ Wu,
                                               __bf16* __restrict__ Wu_t)
{
    int idx = blockIdx.x * 256 + threadIdx.x;       // 524288 total
    int kk = idx & 511, n = idx >> 9;               // Wu flat: kk*1024 + n
    Wu_t[idx] = to_bf16(Wu[(size_t)kk * D_MODEL + n]);
}

// ---------------------------------------------------------------------------
// Kernel 1: gating. One wave per token; top-2 softmax scaled by 0.5.
// ---------------------------------------------------------------------------
__global__ __launch_bounds__(256) void gate_kernel(const float* __restrict__ x,
                                                   const float* __restrict__ wg,
                                                   float* __restrict__ gates)
{
    const int wv   = threadIdx.x >> 5;
    const int lane = threadIdx.x & 31;
    const int t    = blockIdx.x * 8 + wv;
    const float* xr = x + (size_t)t * D_MODEL;

    float acc[8];
    #pragma unroll
    for (int e = 0; e < 8; ++e) acc[e] = 0.f;

    for (int d = lane; d < D_MODEL; d += 32) {
        float xv = xr[d];
        const float* wr = wg + d * 8;
        v4f w0 = *(const v4f*)wr;
        v4f w1 = *(const v4f*)(wr + 4);
        acc[0] += xv * w0.x; acc[1] += xv * w0.y;
        acc[2] += xv * w0.z; acc[3] += xv * w0.w;
        acc[4] += xv * w1.x; acc[5] += xv * w1.y;
        acc[6] += xv * w1.z; acc[7] += xv * w1.w;
    }
    #pragma unroll
    for (int m = 16; m >= 1; m >>= 1) {
        #pragma unroll
        for (int e = 0; e < 8; ++e)
            acc[e] += __shfl_xor(acc[e], m, 32);
    }
    if (lane == 0) {
        float m1 = -1e30f, m2 = -1e30f; int i1 = 0, i2 = 0;
        #pragma unroll
        for (int e = 0; e < 8; ++e) {
            float v = acc[e];
            if (v > m1)      { m2 = m1; i2 = i1; m1 = v; i1 = e; }
            else if (v > m2) { m2 = v;  i2 = e; }
        }
        float e2  = expf(m2 - m1);
        float inv = 1.f / (1.f + e2);
        float g1  = inv      * ADAPTER_SCALE_F;
        float g2  = e2 * inv * ADAPTER_SCALE_F;
        #pragma unroll
        for (int e = 0; e < 8; ++e)
            gates[(size_t)t * 8 + e] = (e == i1) ? g1 : ((e == i2) ? g2 : 0.f);
    }
}

// ---------------------------------------------------------------------------
// Kernel 2: dense MoE adapter. 4 waves/block (128 thr), one 16-token tile per
// wave -> 64 tokens/block, 128 blocks; ~136 KB LDS so 2 workgroups per WGP.
// Phase A: Hg = gate*0.5*relu(x@Wd_e + bd_e) -> LDS (bf16), all 8 experts.
// Phase B: out = Hg(16x512) @ Wu_all(512x1024) + G(16x8)@bu(8x1024),
//          32 output columns (two 16x16 C tiles) per iteration.
// Weight tiles arrive in LDS via double-buffered async global->LDS copies.
// ---------------------------------------------------------------------------
__global__ __launch_bounds__(128) void moe_kernel(const float*  __restrict__ x,
                                                  const __bf16* __restrict__ Wd_t,
                                                  const float*  __restrict__ bd,
                                                  const __bf16* __restrict__ Wu_t,
                                                  const float*  __restrict__ bu,
                                                  const float*  __restrict__ gates,
                                                  float* __restrict__ out)
{
    extern __shared__ char smem[];
    __bf16* hbuf = (__bf16*)smem;                        // 4*16*512 bf16 =  64 KB
    __bf16* stgA = (__bf16*)(smem + 4 * 16 * 512 * 2);   // 2 x 2048 bf16 =   8 KB
    __bf16* stgB = stgA + 2 * 2048;                      // 2 x 16384 bf16 = 64 KB

    const int tid  = threadIdx.x;
    const int wv   = tid >> 5;
    const int lane = tid & 31;
    const int lmod = lane & 15;   // row (A) or column (B, C) index within tile
    const int lhi  = lane >> 4;   // wave half
    const int t0   = blockIdx.x * 64 + wv * 16;

    // Per-thread async staging slices (uniform across all 128 threads).
    const int an = tid >> 1, ap = (tid & 1) * 16;  // A: 64 rows x 2 x 32B
    const int bn = tid >> 2, bs = (tid & 3) * 128; // B: 32 rows x 4 x 256B

    // ---------------- Phase A ----------------
    v8f hacc[4];
    #pragma unroll
    for (int nt = 0; nt < 4; ++nt)
        #pragma unroll
        for (int j = 0; j < 8; ++j) hacc[nt][j] = 0.f;

    // Prologue: stage chunk 0 (expert 0, kc 0).
    {
        unsigned dst = lds_off(stgA + an * 32 + ap);
        const __bf16* src = Wd_t + (size_t)an * 1024 + ap;
        async_copy_b128(dst,      src);
        async_copy_b128(dst + 16, src + 8);
    }

    for (int ekc = 0; ekc < N_EXPERTS * 32; ++ekc) {
        const int e  = ekc >> 5;
        const int kc = ekc & 31;
        const __bf16* bufA = stgA + (ekc & 1) * 2048;

        wait_async0();          // my async writes into bufA complete
        __syncthreads();        // everyone's writes complete; prev buffer free

        if (ekc + 1 < N_EXPERTS * 32) {
            const int e1 = (ekc + 1) >> 5, kc1 = (ekc + 1) & 31;
            unsigned dst = lds_off(stgA + ((ekc + 1) & 1) * 2048 + an * 32 + ap);
            const __bf16* src = Wd_t + ((size_t)e1 * 64 + an) * 1024 + kc1 * 32 + ap;
            async_copy_b128(dst,      src);
            async_copy_b128(dst + 16, src + 8);
        }

        // A tile: 16 tokens x 32 K of x -> bf16 (ISA 16-bit A layout:
        // lane<16 holds K {0..7,16..23}, lane>=16 holds K {8..15,24..31}).
        const float* xr = x + (size_t)(t0 + lmod) * D_MODEL + kc * 32 + lhi * 8;
        v4f xa0 = *(const v4f*)(xr);
        v4f xa1 = *(const v4f*)(xr + 4);
        v4f xb0 = *(const v4f*)(xr + 16);
        v4f xb1 = *(const v4f*)(xr + 20);
        if (kc + 1 < 32) __builtin_prefetch(xr + 32, 0, 3);
        v16bf a;
        #pragma unroll
        for (int j = 0; j < 4; ++j) {
            a[j]      = to_bf16(xa0[j]);
            a[4 + j]  = to_bf16(xa1[j]);
            a[8 + j]  = to_bf16(xb0[j]);
            a[12 + j] = to_bf16(xb1[j]);
        }

        // B tiles from LDS: lane holds column lmod, same K order as A.
        #pragma unroll
        for (int nt = 0; nt < 4; ++nt) {
            v8bf blo = *(const v8bf*)(bufA + (nt * 16 + lmod) * 32 + lhi * 8);
            v8bf bhi = *(const v8bf*)(bufA + (nt * 16 + lmod) * 32 + 16 + lhi * 8);
            v16bf b = __builtin_shufflevector(blo, bhi,
                0, 1, 2, 3, 4, 5, 6, 7, 8, 9, 10, 11, 12, 13, 14, 15);
            hacc[nt] = __builtin_amdgcn_wmma_f32_16x16x32_bf16(
                false, a, false, b, (short)0, hacc[nt], false, false);
        }

        if (kc == 31) {
            // Epilogue: +bd, relu, fold gate*0.5, store Hg rows (bf16, own rows).
            float grow[8];
            #pragma unroll
            for (int i = 0; i < 8; ++i)
                grow[i] = gates[(size_t)(t0 + i + 8 * lhi) * N_EXPERTS + e];
            #pragma unroll
            for (int nt = 0; nt < 4; ++nt) {
                int col = nt * 16 + lmod;
                float bdv = bd[e * BN + col];
                #pragma unroll
                for (int i = 0; i < 8; ++i) {
                    float v = hacc[nt][i] + bdv;
                    v = v > 0.f ? v : 0.f;
                    v *= grow[i];
                    hbuf[(size_t)(wv * 16 + i + 8 * lhi) * 512 + e * BN + col]
                        = to_bf16(v);
                }
                #pragma unroll
                for (int j = 0; j < 8; ++j) hacc[nt][j] = 0.f;
            }
        }
    }

    // Gate A-operands for the exact-f32 bias WMMA (G 16x8 -> two 16x4 tiles).
    v2f ga0, ga1;
    {
        const float* gp = gates + (size_t)(t0 + lmod) * N_EXPERTS + lhi * 2;
        ga0.x = gp[0]; ga0.y = gp[1];
        ga1.x = gp[4]; ga1.y = gp[5];
    }

    // ---------------- Phase B ----------------
    // Prologue: stage Wu_all[:, 0..31]  (32 rows of Wu_t, 1 KB each).
    {
        unsigned dst = lds_off(stgB + bn * 512 + bs);
        const __bf16* src = Wu_t + (size_t)bn * 512 + bs;
        #pragma unroll
        for (int j = 0; j < 16; ++j)
            async_copy_b128(dst + j * 16, src + j * 8);
    }

    const __bf16* hrow = hbuf + (size_t)(wv * 16 + lmod) * 512;
    for (int nbt = 0; nbt < 32; ++nbt) {
        const int nb = nbt * 32;
        const __bf16* bufB = stgB + (nbt & 1) * 16384;

        wait_async0();
        __syncthreads();

        if (nbt + 1 < 32) {
            unsigned dst = lds_off(stgB + ((nbt + 1) & 1) * 16384 + bn * 512 + bs);
            const __bf16* src = Wu_t + (size_t)((nbt + 1) * 32 + bn) * 512 + bs;
            #pragma unroll
            for (int j = 0; j < 16; ++j)
                async_copy_b128(dst + j * 16, src + j * 8);
        }

        const int col0 = nb + lmod;
        const int col1 = nb + 16 + lmod;
        v8f acc0 = {}, acc1 = {};
#if __has_builtin(__builtin_amdgcn_wmma_f32_16x16x4_f32)
        // bias: out_bias = G(16x8) @ bu(8x1024) via exact-f32 WMMAs.
        v2f b00, b01, b10, b11;
        b00.x = bu[(size_t)(lhi * 2 + 0) * D_MODEL + col0];
        b00.y = bu[(size_t)(lhi * 2 + 1) * D_MODEL + col0];
        b01.x = bu[(size_t)(4 + lhi * 2 + 0) * D_MODEL + col0];
        b01.y = bu[(size_t)(4 + lhi * 2 + 1) * D_MODEL + col0];
        b10.x = bu[(size_t)(lhi * 2 + 0) * D_MODEL + col1];
        b10.y = bu[(size_t)(lhi * 2 + 1) * D_MODEL + col1];
        b11.x = bu[(size_t)(4 + lhi * 2 + 0) * D_MODEL + col1];
        b11.y = bu[(size_t)(4 + lhi * 2 + 1) * D_MODEL + col1];
        acc0 = __builtin_amdgcn_wmma_f32_16x16x4_f32(
            false, ga0, false, b00, (short)0, acc0, false, false);
        acc0 = __builtin_amdgcn_wmma_f32_16x16x4_f32(
            false, ga1, false, b01, (short)0, acc0, false, false);
        acc1 = __builtin_amdgcn_wmma_f32_16x16x4_f32(
            false, ga0, false, b10, (short)0, acc1, false, false);
        acc1 = __builtin_amdgcn_wmma_f32_16x16x4_f32(
            false, ga1, false, b11, (short)0, acc1, false, false);
#else
        #pragma unroll
        for (int i = 0; i < 8; ++i) {
            float s0 = 0.f, s1 = 0.f;
            #pragma unroll
            for (int e = 0; e < 8; ++e) {
                float g = gates[(size_t)(t0 + i + 8 * lhi) * N_EXPERTS + e];
                s0 += g * bu[(size_t)e * D_MODEL + col0];
                s1 += g * bu[(size_t)e * D_MODEL + col1];
            }
            acc0[i] = s0; acc1[i] = s1;
        }
#endif
        // Main GEMM: K = 512 in 16 bf16 WMMA steps; A fragment reused for
        // both column tiles.
        #pragma unroll
        for (int kc = 0; kc < 16; ++kc) {
            v8bf lo = *(const v8bf*)(hrow + kc * 32 + lhi * 8);
            v8bf hi = *(const v8bf*)(hrow + kc * 32 + 16 + lhi * 8);
            v16bf a = __builtin_shufflevector(lo, hi,
                0, 1, 2, 3, 4, 5, 6, 7, 8, 9, 10, 11, 12, 13, 14, 15);
            v8bf b0l = *(const v8bf*)(bufB + lmod * 512 + kc * 32 + lhi * 8);
            v8bf b0h = *(const v8bf*)(bufB + lmod * 512 + kc * 32 + 16 + lhi * 8);
            v8bf b1l = *(const v8bf*)(bufB + (16 + lmod) * 512 + kc * 32 + lhi * 8);
            v8bf b1h = *(const v8bf*)(bufB + (16 + lmod) * 512 + kc * 32 + 16 + lhi * 8);
            v16bf b0 = __builtin_shufflevector(b0l, b0h,
                0, 1, 2, 3, 4, 5, 6, 7, 8, 9, 10, 11, 12, 13, 14, 15);
            v16bf b1 = __builtin_shufflevector(b1l, b1h,
                0, 1, 2, 3, 4, 5, 6, 7, 8, 9, 10, 11, 12, 13, 14, 15);
            acc0 = __builtin_amdgcn_wmma_f32_16x16x32_bf16(
                false, a, false, b0, (short)0, acc0, false, false);
            acc1 = __builtin_amdgcn_wmma_f32_16x16x32_bf16(
                false, a, false, b1, (short)0, acc1, false, false);
        }

        // Write f32 output (C/D layout: vgpr i -> row i + 8*lhi, column lmod).
        #pragma unroll
        for (int i = 0; i < 8; ++i) {
            out[(size_t)(t0 + i + 8 * lhi) * D_MODEL + col0] = acc0[i];
            out[(size_t)(t0 + i + 8 * lhi) * D_MODEL + col1] = acc1[i];
        }
    }
}

extern "C" void kernel_launch(void* const* d_in, const int* in_sizes, int n_in,
                              void* d_out, int out_size, void* d_ws, size_t ws_size,
                              hipStream_t stream) {
    (void)in_sizes; (void)n_in; (void)out_size; (void)ws_size;
    const float* x      = (const float*)d_in[0];
    const float* w_gate = (const float*)d_in[1];
    // d_in[2] = w_noise (unused in reference forward path)
    const float* Wd     = (const float*)d_in[3];
    const float* bd     = (const float*)d_in[4];
    const float* Wu     = (const float*)d_in[5];
    const float* bu     = (const float*)d_in[6];
    float* out   = (float*)d_out;

    // Workspace layout: gates (256 KB) | Wd_t bf16 (1 MB) | Wu_t bf16 (1 MB).
    float*  gates = (float*)d_ws;
    __bf16* Wd_t  = (__bf16*)((char*)d_ws + 256 * 1024);
    __bf16* Wu_t  = (__bf16*)((char*)d_ws + 256 * 1024 + 1024 * 1024);

    prep_wd<<<2048, 256, 0, stream>>>(Wd, Wd_t);
    prep_wu<<<2048, 256, 0, stream>>>(Wu, Wu_t);
    gate_kernel<<<T_TOKENS / 8, 256, 0, stream>>>(x, w_gate, gates);

    const size_t lds_bytes = (size_t)4 * 16 * 512 * 2   // Hg     ( 64 KB)
                           + (size_t)2 * 2048 * 2       // Wd stg (  8 KB)
                           + (size_t)2 * 16384 * 2;     // Wu stg ( 64 KB)
    moe_kernel<<<T_TOKENS / 64, 128, lds_bytes, stream>>>(
        x, Wd_t, bd, Wu_t, bu, gates, out);
}
